// CRF_23819888623939
// MI455X (gfx1250) — matrix operand order repaired and checked by reading, not compiled, not run on hardware
//
#include <hip/hip_runtime.h>
#include <math.h>

typedef __attribute__((ext_vector_type(16))) _Float16 v16h;
typedef __attribute__((ext_vector_type(8)))  float    v8f;

#define B_SZ 512
#define T_SZ 512
#define C_SZ 64

#define LOG2E 1.44269504088896340736f
#define LN2   0.69314718055994530942f

union HFrag { v16h h; unsigned int u[8]; };
union H2    { _Float16 h[2]; unsigned int u; };
union H8    { _Float16 h[8]; uint4 u4; };

__device__ __forceinline__ float fast_exp(float x) {
    return __builtin_amdgcn_exp2f(x * LOG2E);   // v_exp_f32
}
__device__ __forceinline__ float fast_log(float x) {
    return __builtin_amdgcn_logf(x) * LN2;      // v_log_f32 (+fma), input normal here
}

// Workgroup barrier that only drains the LDS counter (DScnt). Unlike
// __syncthreads(), it does NOT wait for in-flight global loads, so the
// register-pipelined emissions prefetch stays outstanding across the barrier.
__device__ __forceinline__ void lds_barrier() {
    asm volatile("s_wait_dscnt 0x0\n\t"
                 "s_barrier_signal -1\n\t"
                 "s_barrier_wait -1" ::: "memory");
}

// max over 8-lane xor groups; pure-VALU v_permlane16_b32 when available
// (no LDS round trips / dscnt waits), else ds_bpermute via __shfl_xor.
__device__ __forceinline__ float rowmax8(float x) {
#if __has_builtin(__builtin_amdgcn_permlane16)
    unsigned u, p;
    u = __float_as_uint(x);
    p = __builtin_amdgcn_permlane16(u, u, 0x67452301u, 0xEFCDAB89u, false, false); // xor 1
    x = fmaxf(x, __uint_as_float(p));
    u = __float_as_uint(x);
    p = __builtin_amdgcn_permlane16(u, u, 0x54761032u, 0xDCFE98BAu, false, false); // xor 2
    x = fmaxf(x, __uint_as_float(p));
    u = __float_as_uint(x);
    p = __builtin_amdgcn_permlane16(u, u, 0x32107654u, 0xBA98FEDCu, false, false); // xor 4
    x = fmaxf(x, __uint_as_float(p));
#else
    x = fmaxf(x, __shfl_xor(x, 1, 8));
    x = fmaxf(x, __shfl_xor(x, 2, 8));
    x = fmaxf(x, __shfl_xor(x, 4, 8));
#endif
    return x;
}

// ---------------------------------------------------------------------------
// Forward algorithm (log-partition) via WMMA.
// lp'[b,j] = log( sum_i exp(lp[b,i]-m_b) * exp(trans[i,j]) ) + m_b + emit[b,t,j]
// One block = 16 batches, 4 waves; wave w owns output columns [16w, 16w+16).
// exp(transitions) B-fragments are loop-invariant and live in VGPRs.
// Row maxima computed in the A-build phase from the lp row already being
// loaded (max-of-8 + 3 permlane16 steps), republished via rowM[16].
// Whole mask tile prestaged transposed to LDS as 0/1 floats; the masked
// update is a branch-free fma blend. Emissions software-pipelined (b128)
// and kept in flight across the DS-only barriers.
// ---------------------------------------------------------------------------
__global__ __launch_bounds__(128)
void crf_forward_kernel(const float* __restrict__ emissions,
                        const int*   __restrict__ mask,
                        const float* __restrict__ transitions,
                        const float* __restrict__ start_t,
                        const float* __restrict__ end_t,
                        float*       __restrict__ denom)
{
    __shared__ __align__(16) float     lp[16][C_SZ];      // running log-probs (f32)
    __shared__ __align__(16) _Float16  aTile[16][C_SZ];   // exp(lp - m), WMMA A source
    __shared__ __align__(16) float     emitT[16][C_SZ];   // staged emissions tile
    __shared__ __align__(16) float     rowM[16];          // row maxima (this step)
    __shared__ __align__(16) float     maskAll[T_SZ][16]; // transposed mask, 0/1 floats

    const int tid  = threadIdx.x;
    const int lane = tid & 31;
    const int wave = tid >> 5;                 // 0..3
    const int b0   = blockIdx.x * 16;

    // contiguous-8 element ownership: row mr, cols [mc, mc+8)
    const int mr = tid >> 3;
    const int mc = (tid & 7) * 8;
    const float* emRow = emissions + (size_t)(b0 + mr) * T_SZ * C_SZ + mc;

    // ---- B fragments: exp(transitions), columns [16w, 16w+16), K split 32+32.
    // B k order (contiguous, shared with A): lane n(0-15): N=n, VGPR v = K 2v,2v+1; lanes 16-31: K+16.
    HFrag bfrag[2];
    {
        const int n    = lane & 15;
        const int kb   = (lane < 16) ? 0 : 16;
        const int ncol = wave * 16 + n;
        for (int kk = 0; kk < 2; ++kk)
            for (int v = 0; v < 8; ++v) {
                const int k = kk * 32 + kb + 2 * v;
                H2 p;
                p.h[0] = (_Float16)fast_exp(transitions[(k)     * C_SZ + ncol]);
                p.h[1] = (_Float16)fast_exp(transitions[(k + 1) * C_SZ + ncol]);
                bfrag[kk].u[v] = p.u;
            }
    }

    // ---- prestage whole mask tile, transposed, as 0/1 floats (coalesced reads)
    {
        const int prow  = tid >> 3;            // 0..15
        const int tbase = (tid & 7) * 64;      // 64 timesteps per thread
        const int* mrow = mask + (size_t)(b0 + prow) * T_SZ + tbase;
        for (int i = 0; i < 64; i += 4) {
            int4 mv = *(const int4*)(mrow + i);
            maskAll[tbase + i + 0][prow] = mv.x ? 1.0f : 0.0f;
            maskAll[tbase + i + 1][prow] = mv.y ? 1.0f : 0.0f;
            maskAll[tbase + i + 2][prow] = mv.z ? 1.0f : 0.0f;
            maskAll[tbase + i + 3][prow] = mv.w ? 1.0f : 0.0f;
        }
    }

    // ---- lp0 = start + emissions[:,0,:]
    {
        float4 ea = *(const float4*)(emRow + 0);
        float4 eb = *(const float4*)(emRow + 4);
        float4 sa = *(const float4*)(start_t + mc);
        float4 sb = *(const float4*)(start_t + mc + 4);
        float4 la = make_float4(ea.x + sa.x, ea.y + sa.y, ea.z + sa.z, ea.w + sa.w);
        float4 lb = make_float4(eb.x + sb.x, eb.y + sb.y, eb.z + sb.z, eb.w + sb.w);
        *(float4*)&lp[mr][mc]     = la;
        *(float4*)&lp[mr][mc + 4] = lb;
    }

    // ---- pipeline preload for t = 1
    float4 pa = *(const float4*)(emRow + C_SZ);
    float4 pb = *(const float4*)(emRow + C_SZ + 4);
    lds_barrier();

    for (int t = 1; t < T_SZ; ++t) {
        // ================= phase 1 (pre mid-barrier) =================
        // stage this step's emissions tile from pipeline registers
        *(float4*)&emitT[mr][mc]     = pa;
        *(float4*)&emitT[mr][mc + 4] = pb;

        // A = exp(lp - m); row max computed from the lp row being loaded anyway
        {
            float4 la = *(const float4*)&lp[mr][mc];
            float4 lb = *(const float4*)&lp[mr][mc + 4];
            float m = fmaxf(fmaxf(fmaxf(la.x, la.y), fmaxf(la.z, la.w)),
                            fmaxf(fmaxf(lb.x, lb.y), fmaxf(lb.z, lb.w)));
            m = rowmax8(m);                      // identical across the row's 8 threads
            if ((tid & 7) == 0) rowM[mr] = m;    // republish for writeback add-back
            H8 o;
            o.h[0] = (_Float16)__builtin_amdgcn_exp2f((la.x - m) * LOG2E);
            o.h[1] = (_Float16)__builtin_amdgcn_exp2f((la.y - m) * LOG2E);
            o.h[2] = (_Float16)__builtin_amdgcn_exp2f((la.z - m) * LOG2E);
            o.h[3] = (_Float16)__builtin_amdgcn_exp2f((la.w - m) * LOG2E);
            o.h[4] = (_Float16)__builtin_amdgcn_exp2f((lb.x - m) * LOG2E);
            o.h[5] = (_Float16)__builtin_amdgcn_exp2f((lb.y - m) * LOG2E);
            o.h[6] = (_Float16)__builtin_amdgcn_exp2f((lb.z - m) * LOG2E);
            o.h[7] = (_Float16)__builtin_amdgcn_exp2f((lb.w - m) * LOG2E);
            *(uint4*)&aTile[mr][mc] = o.u4;
        }
        lds_barrier();

        // ================= phase 2 (post mid-barrier) =================
        // issue next step's global loads (consumed at next loop top; they stay
        // in flight across the DS-only bottom barrier)
        if (t + 1 < T_SZ) {
            const float* p = emRow + (size_t)(t + 1) * C_SZ;
            pa = *(const float4*)(p);
            pb = *(const float4*)(p + 4);
        }

        const int n     = lane & 15;
        const int col   = wave * 16 + n;
        const int rbase = (lane < 16) ? 0 : 8;

        // A fragments in the same contiguous k order as B (4x ds_load_b128)
        HFrag afrag[2];
        {
            const int row = lane & 15;
            const int kb  = (lane < 16) ? 0 : 16;
            for (int kk = 0; kk < 2; ++kk)
                for (int v = 0; v < 8; ++v) {
                    const int k = kk * 32 + kb + 2 * v;
                    afrag[kk].u[v] = *(const unsigned int*)&aTile[row][k];
                }
        }

        // batched preloads for the writeback (mergeable 2addr/stride64 loads)
        float4 mfa = *(const float4*)&maskAll[t][rbase];
        float4 mfb = *(const float4*)&maskAll[t][rbase + 4];
        float4 rma = *(const float4*)&rowM[rbase];
        float4 rmb = *(const float4*)&rowM[rbase + 4];
        float oldv[8], emv[8];
        #pragma unroll
        for (int r = 0; r < 8; ++r) {
            oldv[r] = lp[rbase + r][col];
            emv[r]  = emitT[rbase + r][col];
        }
        const float mkf[8] = {mfa.x, mfa.y, mfa.z, mfa.w, mfb.x, mfb.y, mfb.z, mfb.w};
        const float rm[8]  = {rma.x, rma.y, rma.z, rma.w, rmb.x, rmb.y, rmb.z, rmb.w};

        // C[16x16] = A[16x64] x B[64x16]  (two K=32 WMMAs)
        v8f acc = {};
        acc = __builtin_amdgcn_wmma_f32_16x16x32_f16(
                  false, afrag[0].h, false, bfrag[0].h, (short)0, acc, false, false);
        acc = __builtin_amdgcn_wmma_f32_16x16x32_f16(
                  false, afrag[1].h, false, bfrag[1].h, (short)0, acc, false, false);

        // lp' = blend(mask, log(acc)+m+emit, old); branch-free, unconditional stores
        #pragma unroll
        for (int r = 0; r < 8; ++r) {
            const float lpn = fast_log(acc[r]) + rm[r] + emv[r];
            const float v   = fmaf(mkf[r], lpn - oldv[r], oldv[r]);
            lp[rbase + r][col] = v;
        }
        lds_barrier();
    }

    // ---- denom[b] = logsumexp(lp[b,:] + end_transitions)
    if (tid < 16) {
        float m = -INFINITY;
        for (int c = 0; c < C_SZ; ++c) m = fmaxf(m, lp[tid][c] + end_t[c]);
        float s = 0.0f;
        for (int c = 0; c < C_SZ; ++c) s += fast_exp(lp[tid][c] + end_t[c] - m);
        denom[b0 + tid] = m + fast_log(s);
    }
}

// ---------------------------------------------------------------------------
// Joint (numerator) score: one thread per batch element.
// ---------------------------------------------------------------------------
__global__ __launch_bounds__(256)
void crf_joint_kernel(const float* __restrict__ emissions,
                      const int*   __restrict__ tags,
                      const int*   __restrict__ mask,
                      const float* __restrict__ transitions,
                      const float* __restrict__ start_t,
                      const float* __restrict__ end_t,
                      float*       __restrict__ numer)
{
    const int b = blockIdx.x * blockDim.x + threadIdx.x;
    if (b >= B_SZ) return;
    const int*   tg = tags + (size_t)b * T_SZ;
    const int*   mk = mask + (size_t)b * T_SZ;
    const float* em = emissions + (size_t)b * T_SZ * C_SZ;

    int   prev  = tg[0];
    float score = start_t[prev] + em[prev];
    int   cnt   = mk[0];
    for (int t = 1; t < T_SZ; ++t) {
        const int cur = tg[t];
        const float sc = transitions[prev * C_SZ + cur] + em[(size_t)t * C_SZ + cur];
        const int m = mk[t];
        score += m ? sc : 0.0f;
        cnt   += m;
        prev   = cur;
    }
    score += end_t[tg[cnt - 1]];
    numer[b] = score;
}

// ---------------------------------------------------------------------------
// mean(denom - numer) over B
// ---------------------------------------------------------------------------
__global__ __launch_bounds__(256)
void crf_reduce_kernel(const float* __restrict__ denom,
                       const float* __restrict__ numer,
                       float*       __restrict__ out)
{
    __shared__ float sh[256];
    const int tid = threadIdx.x;
    float s = (denom[tid]       - numer[tid]) +
              (denom[tid + 256] - numer[tid + 256]);
    sh[tid] = s;
    __syncthreads();
    for (int off = 128; off > 0; off >>= 1) {
        if (tid < off) sh[tid] += sh[tid + off];
        __syncthreads();
    }
    if (tid == 0) out[0] = sh[0] * (1.0f / (float)B_SZ);
}

extern "C" void kernel_launch(void* const* d_in, const int* in_sizes, int n_in,
                              void* d_out, int out_size, void* d_ws, size_t ws_size,
                              hipStream_t stream)
{
    const float* emissions   = (const float*)d_in[0];
    const int*   tags        = (const int*)  d_in[1];
    const int*   mask        = (const int*)  d_in[2];
    const float* transitions = (const float*)d_in[3];
    const float* start_t     = (const float*)d_in[4];
    const float* end_t       = (const float*)d_in[5];

    float* denom = (float*)d_ws;          // 512 floats
    float* numer = denom + B_SZ;          // 512 floats

    crf_forward_kernel<<<B_SZ / 16, 128, 0, stream>>>(
        emissions, mask, transitions, start_t, end_t, denom);
    crf_joint_kernel<<<B_SZ / 256, 256, 0, stream>>>(
        emissions, tags, mask, transitions, start_t, end_t, numer);
    crf_reduce_kernel<<<1, 256, 0, stream>>>(denom, numer, (float*)d_out);
}
